// DiffWaveBlock_75763223101920
// MI455X (gfx1250) — matrix operand order repaired and checked by reading, not compiled, not run on hardware
//
#include <hip/hip_runtime.h>
#include <hip/hip_bf16.h>

// DiffWave S4 block for MI455X (gfx1250, wave32).
//  - FFT convolution done per-row in LDS (packed-real 8192-pt complex FFT,
//    64KB data + 32KB twiddles: exploits CDNA5's 320KB/WGP LDS).
//  - All channel-mixing GEMMs via v_wmma_f32_16x16x32_bf16 with weights
//    pre-swizzled to the ISA fragment layout (CDNA5 ISA 7.12.2).
//  - Tile staging in the fused mix kernel uses GLOBAL_LOAD_ASYNC_TO_LDS_B128
//    (ASYNCcnt path) when the toolchain exposes the builtin.

#define Bb   8
#define Hh   256
#define Lseq 8192
#define Nst  16
#define DTE  512
#define Mfft 8192      // half of conv length 2L
#define LOG2M 13
#define NT   32        // columns per mix workgroup

#if defined(__gfx1250__) && __has_builtin(__builtin_amdgcn_global_load_async_to_lds_b128)
#define USE_ASYNC_LDS 1
#else
#define USE_ASYNC_LDS 0
#endif

typedef __attribute__((ext_vector_type(16))) __bf16 v16bf;
typedef __attribute__((ext_vector_type(8)))  float  v8f;

__device__ __forceinline__ unsigned short f2bf(float f) {
  union { float f; unsigned u; } v; v.f = f;
  unsigned r = v.u + 0x7FFFu + ((v.u >> 16) & 1u);   // RNE truncate to bf16
  return (unsigned short)(r >> 16);
}
__device__ __forceinline__ float gelu_t(float x) {
  const float c = 0.7978845608028654f;
  return 0.5f * x * (1.0f + tanhf(c * (x + 0.044715f * x * x * x)));
}
__device__ __forceinline__ float sigm(float x) { return 1.0f / (1.0f + __expf(-x)); }

// ---------------- small kernels ----------------

__global__ __launch_bounds__(256) void part_t_kernel(
    const float* __restrict__ emb, const float* __restrict__ w,
    const float* __restrict__ bias, float* __restrict__ pt) {
  int id = blockIdx.x * 256 + threadIdx.x;       // B*H = 2048
  int b = id / Hh, h = id % Hh;
  float s = 0.f;
  const float* e = emb + (size_t)b * DTE;
  const float* wr = w + (size_t)h * DTE;
  for (int k = 0; k < DTE; ++k) s += e[k] * wr[k];
  pt[id] = s + bias[h];
}

__global__ __launch_bounds__(256) void tln1_kernel(
    const float* __restrict__ x, const float* __restrict__ m1,
    const float* __restrict__ s1, float* __restrict__ ss) {
  int id = blockIdx.x * 256 + threadIdx.x;       // B*L
  int b = id / Lseq, l = id % Lseq;
  const float* xp = x + (size_t)b * Hh * Lseq + l;
  float sm = 0.f, sq = 0.f;
  for (int h = 0; h < Hh; ++h) { float v = xp[(size_t)h * Lseq]; sm += v; sq += v * v; }
  float mean = sm * (1.0f / Hh);
  float var  = sq * (1.0f / Hh) - mean * mean;
  float stdv = sqrtf(fmaxf(var, 0.f));
  ss[2 * id]     = mean - m1[0];
  ss[2 * id + 1] = s1[0] / stdv;
}

// pre-swizzle weights (MxK f32, row-major) into per-fragment bf16 blocks:
// block fb = mt*Ks+ks holds 32 lanes x 16 bf16; lane: M = mt*16+(lane&15),
// half=lane>>4; element i: p=i>>1,pos=i&1; K = ks*32 + (p<4 ? 2p+pos : 16+2(p-4)+pos) + 8*half
__global__ __launch_bounds__(256) void wswz_kernel(
    const float* __restrict__ src, unsigned short* __restrict__ dst,
    int Ks, int Kdim) {
  int id = blockIdx.x * 256 + threadIdx.x;       // Mdim*Kdim elements
  int fb = id >> 9, r = id & 511;
  int lane = r >> 4, i = r & 15;
  int mt = fb / Ks, ks = fb % Ks;
  int half = lane >> 4, p = i >> 1, pos = i & 1;
  int kk = (p < 4 ? 2 * p + pos : 16 + 2 * (p - 4) + pos) + 8 * half;
  int row = mt * 16 + (lane & 15);
  int col = ks * 32 + kk;
  dst[id] = f2bf(src[(size_t)row * Kdim + col]);
}

// ---------------- FFT helpers (LDS, 256 threads, M=8192) ----------------

__device__ __forceinline__ void fft_dit(float* re, float* im,
                                        const float* twr, const float* twi,
                                        int t, int conjTw) {
  for (int s = 1; s <= LOG2M; ++s) {
    int half = 1 << (s - 1);
    for (int q = 0; q < (Mfft / 2) / 256; ++q) {
      int bt = t + q * 256;
      int jj = bt & (half - 1);
      int i0 = ((bt >> (s - 1)) << s) | jj;
      int i1 = i0 + half;
      int rr = jj << (LOG2M - s);
      float wr = twr[rr], wi = conjTw ? -twi[rr] : twi[rr];
      float xr = re[i1], xi = im[i1];
      float tr = wr * xr - wi * xi;
      float ti = wr * xi + wi * xr;
      float ur = re[i0], ui = im[i0];
      re[i0] = ur + tr; im[i0] = ui + ti;
      re[i1] = ur - tr; im[i1] = ui - ti;
    }
    __syncthreads();
  }
}

__device__ __forceinline__ void fft_dif(float* re, float* im,
                                        const float* twr, const float* twi,
                                        int t, int conjTw) {
  for (int s = LOG2M; s >= 1; --s) {
    int half = 1 << (s - 1);
    for (int q = 0; q < (Mfft / 2) / 256; ++q) {
      int bt = t + q * 256;
      int jj = bt & (half - 1);
      int i0 = ((bt >> (s - 1)) << s) | jj;
      int i1 = i0 + half;
      int rr = jj << (LOG2M - s);
      float wr = twr[rr], wi = conjTw ? -twi[rr] : twi[rr];
      float ur = re[i0], ui = im[i0];
      float vr = re[i1], vi = im[i1];
      re[i0] = ur + vr; im[i0] = ui + vi;
      float dr = ur - vr, di = ui - vi;
      re[i1] = wr * dr - wi * di;
      im[i1] = wr * di + wi * dr;
    }
    __syncthreads();
  }
}

// ---------------- S4D kernel spectra: Kf[h][k], k=0..M ----------------

__global__ __launch_bounds__(256) void kfft_kernel(
    const float* __restrict__ log_dt, const float* __restrict__ A_re,
    const float* __restrict__ A_im, const float* __restrict__ C_re,
    const float* __restrict__ C_im, float* __restrict__ Kf) {
  __shared__ float re[Mfft], im[Mfft];
  __shared__ float twr[Mfft / 2], twi[Mfft / 2];
  __shared__ float ctre[2][Nst], ctim[2][Nst], dar[Nst], dai[Nst];
  int h = blockIdx.x, t = threadIdx.x;

  if (t < Nst) {
    float dt = __expf(log_dt[h]);
    float ar = A_re[h * Nst + t], ai = A_im[h * Nst + t];
    float dr = ar * dt, di = ai * dt;
    dar[t] = dr; dai[t] = di;
    float er = __expf(dr), sn, cs;
    __sincosf(di, &sn, &cs);
    float nr = er * cs - 1.0f, ni = er * sn;          // exp(dtA)-1
    float den = ar * ar + ai * ai;
    float qr = (nr * ar + ni * ai) / den, qi = (ni * ar - nr * ai) / den;
    for (int c = 0; c < 2; ++c) {
      float cr = C_re[((size_t)c * Hh + h) * Nst + t];
      float ci = C_im[((size_t)c * Hh + h) * Nst + t];
      ctre[c][t] = cr * qr - ci * qi;
      ctim[c][t] = cr * qi + ci * qr;
    }
  }
  for (int r = t; r < Mfft / 2; r += 256) {
    float s_, c_;
    __sincosf(-6.283185307179586f * (float)r / (float)Mfft, &s_, &c_);
    twr[r] = c_; twi[r] = s_;
  }
  __syncthreads();

  // generate packed time series with bit-reversed store
  for (int i = 0; i < Mfft / 256; ++i) {
    int j = t + i * 256;
    float v[2];
    for (int half = 0; half < 2; ++half) {
      int tau = 2 * j + half;
      int c, tt;
      if (tau < Lseq) { c = 0; tt = tau; } else { c = 1; tt = 2 * Lseq - 1 - tau; }
      float acc = 0.f;
      for (int n = 0; n < Nst; ++n) {
        float er = __expf(dar[n] * (float)tt);
        float sn, cs;
        sincosf(dai[n] * (float)tt, &sn, &cs);       // accurate reduction (large args)
        acc += er * (ctre[c][n] * cs - ctim[c][n] * sn);
      }
      v[half] = 2.0f * acc;
    }
    int jr = __brev((unsigned)j) >> (32 - LOG2M);
    re[jr] = v[0]; im[jr] = v[1];
  }
  __syncthreads();
  fft_dit(re, im, twr, twi, t, 0);

  // unpack packed-real spectrum -> Kf row (k = 0..M)
  float* G = Kf + (size_t)h * (Mfft + 1) * 2;
  for (int i = 0; i < 16; ++i) {
    int k = 1 + t + i * 256;
    if (k > Mfft / 2) continue;
    float zkr = re[k], zki = im[k], zmr = re[Mfft - k], zmi = im[Mfft - k];
    float zer = 0.5f * (zkr + zmr), zei = 0.5f * (zki - zmi);
    float zor = 0.5f * (zki + zmi), zoi = -0.5f * (zkr - zmr);
    float sn, cs;
    __sincosf(-3.14159265358979f * (float)k / (float)Mfft, &sn, &cs);
    float wzr = cs * zor - sn * zoi, wzi = cs * zoi + sn * zor;
    G[2 * k] = zer + wzr;            G[2 * k + 1] = zei + wzi;
    G[2 * (Mfft - k)] = zer - wzr;   G[2 * (Mfft - k) + 1] = -(zei - wzi);
  }
  if (t == 0) {
    float z0r = re[0], z0i = im[0];
    G[0] = z0r + z0i;            G[1] = 0.f;
    G[2 * Mfft] = z0r - z0i;     G[2 * Mfft + 1] = 0.f;
  }
}

// ---------------- per-row FFT convolution + gelu -> bf16 yconv ----------------

__global__ __launch_bounds__(256) void conv_kernel(
    const float* __restrict__ x, const float* __restrict__ ss,
    const float* __restrict__ pt, const float* __restrict__ Kf,
    const float* __restrict__ Dv, unsigned short* __restrict__ yc) {
  __shared__ float re[Mfft], im[Mfft];
  __shared__ float twr[Mfft / 2], twi[Mfft / 2];
  int bh = blockIdx.x, t = threadIdx.x;
  int b = bh / Hh, h = bh % Hh;
  const float* xrow = x + (size_t)bh * Lseq;
  const float* ssr = ss + (size_t)b * Lseq * 2;
  float ptv = pt[b * Hh + h], Dh = Dv[h];

  for (int r = t; r < Mfft / 2; r += 256) {
    float s_, c_;
    __sincosf(-6.283185307179586f * (float)r / (float)Mfft, &s_, &c_);
    twr[r] = c_; twi[r] = s_;
  }
  // packed load of u' = TLN(x)+part_t, zero-padded to 2L; bit-reversed store
  for (int i = 0; i < Mfft / 256; ++i) {
    int j = t + i * 256;
    float vr = 0.f, vi = 0.f;
    if (j < Lseq / 2) {
      int t0 = 2 * j;
      vr = (xrow[t0]     - ssr[2 * t0])     * ssr[2 * t0 + 1] + ptv;
      vi = (xrow[t0 + 1] - ssr[2 * t0 + 2]) * ssr[2 * t0 + 3] + ptv;
    }
    int jr = __brev((unsigned)j) >> (32 - LOG2M);
    re[jr] = vr; im[jr] = vi;
  }
  __syncthreads();
  fft_dit(re, im, twr, twi, t, 0);

  // unpack U, Y=U*G, repack Zy (each pair owned by one thread)
  const float* G = Kf + (size_t)h * (Mfft + 1) * 2;
  for (int i = 0; i < 16; ++i) {
    int k = 1 + t + i * 256;
    if (k > Mfft / 2) continue;
    float zkr = re[k], zki = im[k], zmr = re[Mfft - k], zmi = im[Mfft - k];
    float zer = 0.5f * (zkr + zmr), zei = 0.5f * (zki - zmi);
    float zor = 0.5f * (zki + zmi), zoi = -0.5f * (zkr - zmr);
    float sn, cs;
    __sincosf(3.14159265358979f * (float)k / (float)Mfft, &sn, &cs); // e^{+i pi k/M}
    float wzr = cs * zor + sn * zoi, wzi = cs * zoi - sn * zor;      // W^k*Zo (W=e^{-i..})
    float ukr = zer + wzr,  uki = zei + wzi;                          // U[k]
    float umr = zer - wzr,  umi = -(zei - wzi);                       // U[M-k]
    float gkr = G[2 * k], gki = G[2 * k + 1];
    float gmr = G[2 * (Mfft - k)], gmi = G[2 * (Mfft - k) + 1];
    float ykr = ukr * gkr - uki * gki, yki = ukr * gki + uki * gkr;
    float ymr = umr * gmr - umi * gmi, ymi = umr * gmi + umi * gmr;
    float yer = 0.5f * (ykr + ymr), yei = 0.5f * (yki - ymi);
    float bcr = ykr - ymr, bci = yki + ymi;
    float qr = 0.5f * (cs * bcr - sn * bci);   // Zo_y = 0.5*conj(W^k)*Bc
    float qi = 0.5f * (cs * bci + sn * bcr);
    re[k] = yer - qi;          im[k] = yei + qr;          // Zy[k]
    re[Mfft - k] = yer + qi;   im[Mfft - k] = -yei + qr;  // Zy[M-k]
  }
  if (t == 0) {
    float z0r = re[0], z0i = im[0];
    float y0 = (z0r + z0i) * G[0];
    float yM = (z0r - z0i) * G[2 * Mfft];
    re[0] = 0.5f * (y0 + yM); im[0] = 0.5f * (y0 - yM);
  }
  __syncthreads();
  fft_dif(re, im, twr, twi, t, 1);   // inverse (unscaled), output bit-reversed

  unsigned int* ycrow = (unsigned int*)(yc + (size_t)bh * Lseq);
  const float inv = 1.0f / (float)Mfft;
  for (int i = 0; i < 16; ++i) {
    int j = t + i * 256;                         // output pair index, j < L/2
    int jr = __brev((unsigned)j) >> (32 - LOG2M);
    float yr = re[jr] * inv, yi = im[jr] * inv;
    int t0 = 2 * j;
    float u0 = (xrow[t0]     - ssr[2 * t0])     * ssr[2 * t0 + 1] + ptv;
    float u1 = (xrow[t0 + 1] - ssr[2 * t0 + 2]) * ssr[2 * t0 + 3] + ptv;
    unsigned a0 = f2bf(gelu_t(yr + u0 * Dh));
    unsigned a1 = f2bf(gelu_t(yi + u1 * Dh));
    ycrow[j] = a0 | (a1 << 16);
  }
}

// ---------------- fused GLU-proj + residual + TLN2 + FFN (WMMA) ----------------

__global__ __launch_bounds__(256) void mix_kernel(
    const unsigned short* __restrict__ yc, const float* __restrict__ x,
    const unsigned short* __restrict__ wA, const float* __restrict__ outb,
    const unsigned short* __restrict__ w1, const float* __restrict__ fb1,
    const unsigned short* __restrict__ w2, const float* __restrict__ fb2,
    const float* __restrict__ n2m, const float* __restrict__ n2s,
    float* __restrict__ out) {
  __shared__ alignas(32) char smem[107520];
  unsigned short* ytile = (unsigned short*)smem;               // 256 x 48 bf16 (reused: y2)
  float* zbuf = (float*)(smem + 24576);                        // 512 x 33 f32 (rows 0..255 -> r1)
  unsigned short* hbuf = (unsigned short*)(smem + 24576 + 33792); // 512 x 48 bf16 (overlaps dead Z-hi)
  __shared__ float pmean[256], pvar[256], colsh[NT], colsc[NT];

  const int t = threadIdx.x, lane = t & 31, w = t >> 5;
  const int g = blockIdx.x;
  const int bI = g / (Lseq / NT);
  const int l0 = (g % (Lseq / NT)) * NT;

  const unsigned short* ycb = yc + (size_t)bI * Hh * Lseq;
#if USE_ASYNC_LDS
  {
    // Stage 256x32 bf16 tile via async LDS DMA: 1024 chunks of 16B.
    // Builtin signature (from diagnostics): (v4i __device__*, v4i __shared__*, Ii, Ii).
    // LDS addrspace(3) offset = low 32 bits of the flat pointer (ISA 10.2).
    typedef int v4i_ __attribute__((vector_size(16)));
    typedef __attribute__((address_space(1))) v4i_ gvec_t;
    typedef __attribute__((address_space(3))) v4i_ lvec_t;
    for (int i = 0; i < 4; ++i) {
      int idx = t + i * 256;
      int row = idx >> 2, ch = idx & 3;            // 4x16B per 64B row
      gvec_t* gp = (gvec_t*)(uintptr_t)(ycb + (size_t)row * Lseq + l0 + ch * 8);
      lvec_t* lp = (lvec_t*)(unsigned)(uintptr_t)(void*)&ytile[row * 48 + ch * 8];
      __builtin_amdgcn_global_load_async_to_lds_b128(gp, lp, 0, 0);
    }
#if __has_builtin(__builtin_amdgcn_s_wait_asynccnt)
    __builtin_amdgcn_s_wait_asynccnt(0);
#else
    asm volatile("s_wait_asynccnt 0x0" ::: "memory");
#endif
  }
#else
  for (int i = 0; i < 32; ++i) {
    int idx = t + i * 256;
    int hh = idx >> 5, c = idx & 31;
    ytile[hh * 48 + c] = ycb[(size_t)hh * Lseq + l0 + c];
  }
#endif
  __syncthreads();

  // GEMM1: Z = out_w(512x256) * ytile + out_b
  {
    const v16bf* Af = (const v16bf*)wA;
    for (int mt = w; mt < 32; mt += 8) {
      v16bf a[8];
#pragma unroll
      for (int ks = 0; ks < 8; ++ks) a[ks] = Af[(mt * 8 + ks) * 32 + lane];
#pragma unroll
      for (int nt = 0; nt < 2; ++nt) {
        v8f acc = {0.f, 0.f, 0.f, 0.f, 0.f, 0.f, 0.f, 0.f};
#pragma unroll
        for (int ks = 0; ks < 8; ++ks) {
          v16bf bb;
          for (int s = 0; s < 16; ++s) bb[s] = ((const __bf16*)ytile)[(ks * 32 + (s < 8 ? 0 : 16) + ((lane >> 4) << 3) + (s & 7)) * 48 + nt * 16 + (lane & 15)];
          acc = __builtin_amdgcn_wmma_f32_16x16x32_bf16(false, a[ks], false, bb,
                                                        (short)0, acc, false, false);
        }
        int col = nt * 16 + (lane & 15);
        int ob = mt * 16 + ((lane >> 4) << 3);
#pragma unroll
        for (int e = 0; e < 8; ++e) zbuf[(ob + e) * 33 + col] = acc[e] + outb[ob + e];
      }
    }
  }
  __syncthreads();

  // GLU + residual -> r1 in zbuf rows 0..255
  const float* xb = x + (size_t)bI * Hh * Lseq;
  for (int i = 0; i < 32; ++i) {
    int idx = t + i * 256;
    int o = idx >> 5, c = idx & 31;
    float za = zbuf[o * 33 + c], zb = zbuf[(o + 256) * 33 + c];
    zbuf[o * 33 + c] = xb[(size_t)o * Lseq + l0 + c] + za * sigm(zb);
  }
  __syncthreads();

  // TLN2 over 256 channels per column
  {
    int c = t & 31, part = t >> 5;
    float sm = 0.f, sq = 0.f;
    for (int o = part * 32; o < part * 32 + 32; ++o) {
      float v = zbuf[o * 33 + c]; sm += v; sq += v * v;
    }
    pmean[part * 32 + c] = sm; pvar[part * 32 + c] = sq;
  }
  __syncthreads();
  if (t < 32) {
    float sm = 0.f, sq = 0.f;
    for (int p = 0; p < 8; ++p) { sm += pmean[p * 32 + t]; sq += pvar[p * 32 + t]; }
    float mean = sm * (1.0f / Hh);
    float var = sq * (1.0f / Hh) - mean * mean;
    colsh[t] = mean - n2m[0];
    colsc[t] = n2s[0] / sqrtf(fmaxf(var, 0.f));
  }
  __syncthreads();
  for (int i = 0; i < 32; ++i) {
    int idx = t + i * 256;
    int o = idx >> 5, c = idx & 31;
    ytile[o * 48 + c] = f2bf((zbuf[o * 33 + c] - colsh[c]) * colsc[c]);
  }
  __syncthreads();

  // GEMM2: h = gelu(ff_w1(512x256) * y2 + b1) -> bf16 hbuf
  {
    const v16bf* Af = (const v16bf*)w1;
    for (int mt = w; mt < 32; mt += 8) {
      v16bf a[8];
#pragma unroll
      for (int ks = 0; ks < 8; ++ks) a[ks] = Af[(mt * 8 + ks) * 32 + lane];
#pragma unroll
      for (int nt = 0; nt < 2; ++nt) {
        v8f acc = {0.f, 0.f, 0.f, 0.f, 0.f, 0.f, 0.f, 0.f};
#pragma unroll
        for (int ks = 0; ks < 8; ++ks) {
          v16bf bb;
          for (int s = 0; s < 16; ++s) bb[s] = ((const __bf16*)ytile)[(ks * 32 + (s < 8 ? 0 : 16) + ((lane >> 4) << 3) + (s & 7)) * 48 + nt * 16 + (lane & 15)];
          acc = __builtin_amdgcn_wmma_f32_16x16x32_bf16(false, a[ks], false, bb,
                                                        (short)0, acc, false, false);
        }
        int col = nt * 16 + (lane & 15);
        int ob = mt * 16 + ((lane >> 4) << 3);
#pragma unroll
        for (int e = 0; e < 8; ++e)
          hbuf[(ob + e) * 48 + col] = f2bf(gelu_t(acc[e] + fb1[ob + e]));
      }
    }
  }
  __syncthreads();

  // GEMM3: out = r1 + ff_w2(256x512) * h + b2
  {
    const v16bf* Af = (const v16bf*)w2;
    float* ob_base = out + (size_t)bI * Hh * Lseq;
    for (int mt = w; mt < 16; mt += 8) {
#pragma unroll
      for (int nt = 0; nt < 2; ++nt) {
        v8f acc = {0.f, 0.f, 0.f, 0.f, 0.f, 0.f, 0.f, 0.f};
#pragma unroll
        for (int ks = 0; ks < 16; ++ks) {
          v16bf a = Af[(mt * 16 + ks) * 32 + lane];
          v16bf bb;
          for (int s = 0; s < 16; ++s) bb[s] = ((const __bf16*)hbuf)[(ks * 32 + (s < 8 ? 0 : 16) + ((lane >> 4) << 3) + (s & 7)) * 48 + nt * 16 + (lane & 15)];
          acc = __builtin_amdgcn_wmma_f32_16x16x32_bf16(false, a, false, bb,
                                                        (short)0, acc, false, false);
        }
        int col = nt * 16 + (lane & 15);
        int ob = mt * 16 + ((lane >> 4) << 3);
#pragma unroll
        for (int e = 0; e < 8; ++e) {
          int o = ob + e;
          ob_base[(size_t)o * Lseq + l0 + col] = zbuf[o * 33 + col] + acc[e] + fb2[o];
        }
      }
    }
  }
}

// ---------------- launcher ----------------

extern "C" void kernel_launch(void* const* d_in, const int* in_sizes, int n_in,
                              void* d_out, int out_size, void* d_ws, size_t ws_size,
                              hipStream_t stream) {
  const float* x      = (const float*)d_in[0];
  const float* emb    = (const float*)d_in[1];
  const float* fc_t_w = (const float*)d_in[2];
  const float* fc_t_b = (const float*)d_in[3];
  const float* log_dt = (const float*)d_in[4];
  const float* A_re   = (const float*)d_in[5];
  const float* A_im   = (const float*)d_in[6];
  const float* C_re   = (const float*)d_in[7];
  const float* C_im   = (const float*)d_in[8];
  const float* Dv     = (const float*)d_in[9];
  const float* out_w  = (const float*)d_in[10];
  const float* out_b  = (const float*)d_in[11];
  const float* n1m    = (const float*)d_in[12];
  const float* n1s    = (const float*)d_in[13];
  const float* n2m    = (const float*)d_in[14];
  const float* n2s    = (const float*)d_in[15];
  const float* ff_w1  = (const float*)d_in[16];
  const float* ff_b1  = (const float*)d_in[17];
  const float* ff_w2  = (const float*)d_in[18];
  const float* ff_b2  = (const float*)d_in[19];

  char* ws = (char*)d_ws;
  const size_t OFF_PT = 0;                       // 8 KB
  const size_t OFF_SS = 8192;                    // 512 KB
  const size_t OFF_KF = 532480;                  // 16.8 MB
  const size_t OFF_YC = 17311744;                // 33.6 MB
  const size_t OFF_WA = 50866176;                // 256 KB
  const size_t OFF_W1 = 51128320;                // 256 KB
  const size_t OFF_W2 = 51390464;                // 256 KB (total ~51.7 MB)

  float* pt = (float*)(ws + OFF_PT);
  float* ss = (float*)(ws + OFF_SS);
  float* Kf = (float*)(ws + OFF_KF);
  unsigned short* yc = (unsigned short*)(ws + OFF_YC);
  unsigned short* wA = (unsigned short*)(ws + OFF_WA);
  unsigned short* w1 = (unsigned short*)(ws + OFF_W1);
  unsigned short* w2 = (unsigned short*)(ws + OFF_W2);

  part_t_kernel<<<(Bb * Hh) / 256, 256, 0, stream>>>(emb, fc_t_w, fc_t_b, pt);
  tln1_kernel<<<(Bb * Lseq) / 256, 256, 0, stream>>>(x, n1m, n1s, ss);
  wswz_kernel<<<(512 * 256) / 256, 256, 0, stream>>>(out_w, wA, 8, 256);
  wswz_kernel<<<(512 * 256) / 256, 256, 0, stream>>>(ff_w1, w1, 8, 256);
  wswz_kernel<<<(256 * 512) / 256, 256, 0, stream>>>(ff_w2, w2, 16, 512);
  kfft_kernel<<<Hh, 256, 0, stream>>>(log_dt, A_re, A_im, C_re, C_im, Kf);
  conv_kernel<<<Bb * Hh, 256, 0, stream>>>(x, ss, pt, Kf, Dv, yc);
  mix_kernel<<<(Bb * Lseq) / NT, 256, 0, stream>>>(yc, x, wA, out_b, w1, ff_b1,
                                                   w2, ff_b2, n2m, n2s,
                                                   (float*)d_out);
  (void)in_sizes; (void)n_in; (void)out_size; (void)ws_size;
}
